// Time_Series_GPT_23287312679525
// MI455X (gfx1250) — matrix operand order repaired and checked by reading, not compiled, not run on hardware
//
#include <hip/hip_runtime.h>
#include <hip/hip_bf16.h>

// ---------------------------------------------------------------------------
// Time-series GPT forward for MI455X (gfx1250, wave32).
//  - All heavy GEMMs: guard-free f16 "NT" WMMA kernel (A[m][k], B[n][k]),
//    double-buffered 64-deep K staging via GLOBAL_LOAD_ASYNC_TO_LDS_B128
//    (ASYNCcnt), 8x v_wmma_f32_16x16x32_f16 per barrier pair, branch-free
//    steady-state loop (last iteration peeled).
//  - Weights pre-transposed+converted to f16 once; activations kept f16.
//  - Guarded f32 WMMA kernel only for the tiny input-MLP / head shapes.
// ---------------------------------------------------------------------------

typedef __attribute__((ext_vector_type(16))) _Float16 v16h;
typedef __attribute__((ext_vector_type(8)))  float    v8f;
typedef __attribute__((ext_vector_type(4)))  int      v4i;

#define BM 128
#define BN 64
#define BK 32          // WMMA K-step
#define BK2 64         // staged K per LDS buffer (2 WMMA k-steps)
#define NTHR 256

#if defined(__has_builtin)
#if __has_builtin(__builtin_amdgcn_global_load_async_to_lds_b128)
#define ASYNC_LDS 1
#endif
#endif

union Frag16 { v16h v; _Float16 h[16]; unsigned u[8]; };
union Frag8f { v8f  v; float f[8]; };
union Pack8h { uint4 q; _Float16 h[8]; };

__device__ __forceinline__ float gelu_exact(float x) {
    return 0.5f * x * (1.0f + erff(x * 0.70710678118654752f));
}

__device__ __forceinline__ void lds_cp16(const _Float16* g, _Float16* l) {
#ifdef ASYNC_LDS
    __builtin_amdgcn_global_load_async_to_lds_b128(
        (__attribute__((address_space(1))) v4i*)g,
        (__attribute__((address_space(3))) v4i*)l, 0, 0);
#else
    *(uint4*)l = *(const uint4*)g;
#endif
}
__device__ __forceinline__ void wait_async_le6() {
#ifdef ASYNC_LDS
#if __has_builtin(__builtin_amdgcn_s_wait_asynccnt)
    __builtin_amdgcn_s_wait_asynccnt(6);
#else
    asm volatile("s_wait_asynccnt 0x6" ::: "memory");
#endif
#endif
}
__device__ __forceinline__ void wait_async_le0() {
#ifdef ASYNC_LDS
#if __has_builtin(__builtin_amdgcn_s_wait_asynccnt)
    __builtin_amdgcn_s_wait_asynccnt(0);
#else
    asm volatile("s_wait_asynccnt 0x0" ::: "memory");
#endif
#endif
}

// stage (BM x BK2) A tile + (BN x BK2) B tile: 16B/lane, 6 async ops per wave
__device__ __forceinline__ void stage_tile_f16(const _Float16* Ag, const _Float16* Bg,
                                               int lda, int ldb, int k0, int tid,
                                               _Float16* As, _Float16* Bs)
{
    int r  = tid >> 3;            // 0..31
    int c8 = (tid & 7) * 8;       // 0..56
    lds_cp16(Ag + (size_t)(r     ) * lda + k0 + c8, As + (r     ) * BK2 + c8);
    lds_cp16(Ag + (size_t)(r + 32) * lda + k0 + c8, As + (r + 32) * BK2 + c8);
    lds_cp16(Ag + (size_t)(r + 64) * lda + k0 + c8, As + (r + 64) * BK2 + c8);
    lds_cp16(Ag + (size_t)(r + 96) * lda + k0 + c8, As + (r + 96) * BK2 + c8);
    lds_cp16(Bg + (size_t)(r     ) * ldb + k0 + c8, Bs + (r     ) * BK2 + c8);
    lds_cp16(Bg + (size_t)(r + 32) * ldb + k0 + c8, Bs + (r + 32) * BK2 + c8);
}

// 2 K-steps (8 WMMAs) from one staged buffer
__device__ __forceinline__ void compute_tile(const _Float16* as, const _Float16* bs,
                                             int wr, int wc, int hw, int l15,
                                             v8f& a00, v8f& a01, v8f& a10, v8f& a11)
{
    #pragma unroll
    for (int ks = 0; ks < 2; ++ks) {
        Frag16 fa0, fa1, fb0, fb1;
        int m0 = wr + l15, m1 = m0 + 16;
        int n0 = wc + l15, n1 = n0 + 16;
        int kb0 = ks * BK;
        #pragma unroll
        for (int j = 0; j < 8; ++j) {
            int ka = kb0 + 16 * hw + 2 * j;
            int kb = kb0 + 16 * hw + 2 * j;
            fa0.u[j] = *(const unsigned*)&as[m0 * BK2 + ka];
            fa1.u[j] = *(const unsigned*)&as[m1 * BK2 + ka];
            fb0.u[j] = *(const unsigned*)&bs[n0 * BK2 + kb];
            fb1.u[j] = *(const unsigned*)&bs[n1 * BK2 + kb];
        }
        a00 = __builtin_amdgcn_wmma_f32_16x16x32_f16(false, fa0.v, false, fb0.v, (short)0, a00, false, false);
        a01 = __builtin_amdgcn_wmma_f32_16x16x32_f16(false, fa0.v, false, fb1.v, (short)0, a01, false, false);
        a10 = __builtin_amdgcn_wmma_f32_16x16x32_f16(false, fa1.v, false, fb0.v, (short)0, a10, false, false);
        a11 = __builtin_amdgcn_wmma_f32_16x16x32_f16(false, fa1.v, false, fb1.v, (short)0, a11, false, false);
    }
}

// ---------------------------------------------------------------------------
// Guard-free f16 NT GEMM: C = act(scale*A@B^T + bias) (+residual)
// A: [M][K] f16 (lda), B: [N][K] f16 (ldb). Output either f32 (Cf, +residual)
// or f16 (Ch); storeCT stores Ch transposed [N][M] (ldct) as packed b128.
// M % 128 == 0, N % 64 == 0, K % 64 == 0 (true for all big GEMMs here).
// ---------------------------------------------------------------------------
__global__ __launch_bounds__(NTHR)
void wmma_gemm_f16nt(const _Float16* __restrict__ A, const _Float16* __restrict__ B,
                     float* __restrict__ Cf, _Float16* __restrict__ Ch,
                     const float* __restrict__ bias, const float* __restrict__ residual,
                     int K, int lda, int ldb, int ldc, int ldct, int storeCT,
                     int act, float scale, int nb1,
                     long long sA0, long long sA1,
                     long long sB0, long long sB1,
                     long long sC0, long long sC1)
{
    int z = blockIdx.z, b0 = z / nb1, b1 = z % nb1;
    A += b0 * sA0 + b1 * sA1;
    B += b0 * sB0 + b1 * sB1;
    long long co = b0 * sC0 + b1 * sC1;
    if (Cf) Cf += co;
    if (Ch) Ch += co;
    if (residual) residual += co;

    __shared__ __align__(16) _Float16 As[2][BM * BK2];
    __shared__ __align__(16) _Float16 Bs[2][BN * BK2];

    const int tid  = threadIdx.x;
    const int lane = tid & 31;
    const int w    = tid >> 5;
    const int wr   = (w & 3) * 32;
    const int wc   = (w >> 2) * 32;
    const int bm   = blockIdx.y * BM;
    const int bn   = blockIdx.x * BN;
    const int hw   = lane >> 4;
    const int l15  = lane & 15;

    const _Float16* Ag = A + (size_t)bm * lda;
    const _Float16* Bg = B + (size_t)bn * ldb;

    v8f a00 = {}, a01 = {}, a10 = {}, a11 = {};
    const int kT2 = K / BK2;

    stage_tile_f16(Ag, Bg, lda, ldb, 0, tid, As[0], Bs[0]);

    // branch-free steady state; last iteration peeled
    for (int kt = 0; kt < kT2 - 1; ++kt) {
        stage_tile_f16(Ag, Bg, lda, ldb, (kt + 1) * BK2, tid,
                       As[(kt + 1) & 1], Bs[(kt + 1) & 1]);
        wait_async_le6();              // previous buffer's 6 async ops complete
        __syncthreads();
        compute_tile(As[kt & 1], Bs[kt & 1], wr, wc, hw, l15, a00, a01, a10, a11);
        __syncthreads();
    }
    wait_async_le0();
    __syncthreads();
    compute_tile(As[(kT2 - 1) & 1], Bs[(kT2 - 1) & 1], wr, wc, hw, l15, a00, a01, a10, a11);

    Frag8f rr[4];
    rr[0].v = a00; rr[1].v = a01; rr[2].v = a10; rr[3].v = a11;
    #pragma unroll
    for (int s = 0; s < 2; ++s) {
        #pragma unroll
        for (int t = 0; t < 2; ++t) {
            Frag8f& fr = rr[s * 2 + t];
            int gmb = bm + wr + s * 16 + 8 * hw;     // rows gmb..gmb+7
            int gn  = bn + wc + t * 16 + l15;
            if (storeCT) {
                Pack8h pk;
                #pragma unroll
                for (int r = 0; r < 8; ++r) pk.h[r] = (_Float16)(fr.f[r] * scale);
                *(uint4*)&Ch[(size_t)gn * ldct + gmb] = pk.q;
            } else if (Ch) {
                #pragma unroll
                for (int r = 0; r < 8; ++r) {
                    float vv = fr.f[r] * scale;
                    if (bias) vv += bias[gn];
                    if (act == 1) vv = fmaxf(vv, 0.0f);
                    else if (act == 2) vv = gelu_exact(vv);
                    Ch[(size_t)(gmb + r) * ldc + gn] = (_Float16)vv;
                }
            } else {
                #pragma unroll
                for (int r = 0; r < 8; ++r) {
                    float vv = fr.f[r] * scale;
                    if (bias) vv += bias[gn];
                    if (act == 1) vv = fmaxf(vv, 0.0f);
                    else if (act == 2) vv = gelu_exact(vv);
                    if (residual) vv += residual[(size_t)(gmb + r) * ldc + gn];
                    Cf[(size_t)(gmb + r) * ldc + gn] = vv;
                }
            }
        }
    }
}

// ---------------------------------------------------------------------------
// Guarded generic f32 GEMM (only for tiny input-MLP / LM-head shapes).
// ---------------------------------------------------------------------------
__global__ __launch_bounds__(NTHR)
void wmma_gemm_f32gen(const float* __restrict__ A, const float* __restrict__ B,
                      float* __restrict__ C, const float* __restrict__ bias,
                      int M, int N, int K, int lda, int ldb, int ldc, int act)
{
    __shared__ __align__(16) _Float16 As[BM * BK];
    __shared__ __align__(16) _Float16 BsT[BN * BK];

    const int tid  = threadIdx.x;
    const int lane = tid & 31;
    const int w    = tid >> 5;
    const int wr   = (w & 3) * 32;
    const int wc   = (w >> 2) * 32;
    const int bm   = blockIdx.y * BM;
    const int bn   = blockIdx.x * BN;
    const int hw   = lane >> 4;
    const int l15  = lane & 15;

    v8f a00 = {}, a01 = {}, a10 = {}, a11 = {};
    const int kTiles = (K + BK - 1) / BK;
    for (int kt = 0; kt < kTiles; ++kt) {
        const int k0g = kt * BK;
        #pragma unroll
        for (int i = 0; i < (BM * BK) / NTHR; ++i) {
            int idx = i * NTHR + tid;
            int r = idx >> 5, c = idx & 31;
            int gm = bm + r, gk = k0g + c;
            float v = (gm < M && gk < K) ? A[(size_t)gm * lda + gk] : 0.0f;
            As[r * BK + c] = (_Float16)v;
        }
        #pragma unroll
        for (int i = 0; i < (BN * BK) / NTHR; ++i) {
            int idx = i * NTHR + tid;
            int nn = idx & 63, kk = idx >> 6;
            int gk = k0g + kk, gn = bn + nn;
            float v = (gk < K && gn < N) ? B[(size_t)gk * ldb + gn] : 0.0f;
            BsT[nn * BK + kk] = (_Float16)v;
        }
        __syncthreads();
        Frag16 fa0, fa1, fb0, fb1;
        {
            int m0 = wr + l15, m1 = m0 + 16;
            int n0 = wc + l15, n1 = n0 + 16;
            #pragma unroll
            for (int j = 0; j < 8; ++j) {
                int ka = 16 * hw + 2 * j;
                int kb = 16 * hw + 2 * j;
                fa0.u[j] = *(const unsigned*)&As[m0 * BK + ka];
                fa1.u[j] = *(const unsigned*)&As[m1 * BK + ka];
                fb0.u[j] = *(const unsigned*)&BsT[n0 * BK + kb];
                fb1.u[j] = *(const unsigned*)&BsT[n1 * BK + kb];
            }
        }
        a00 = __builtin_amdgcn_wmma_f32_16x16x32_f16(false, fa0.v, false, fb0.v, (short)0, a00, false, false);
        a01 = __builtin_amdgcn_wmma_f32_16x16x32_f16(false, fa0.v, false, fb1.v, (short)0, a01, false, false);
        a10 = __builtin_amdgcn_wmma_f32_16x16x32_f16(false, fa1.v, false, fb0.v, (short)0, a10, false, false);
        a11 = __builtin_amdgcn_wmma_f32_16x16x32_f16(false, fa1.v, false, fb1.v, (short)0, a11, false, false);
        __syncthreads();
    }
    Frag8f rr[4];
    rr[0].v = a00; rr[1].v = a01; rr[2].v = a10; rr[3].v = a11;
    #pragma unroll
    for (int s = 0; s < 2; ++s)
        #pragma unroll
        for (int t = 0; t < 2; ++t) {
            Frag8f& fr = rr[s * 2 + t];
            #pragma unroll
            for (int r = 0; r < 8; ++r) {
                int gm = bm + wr + s * 16 + r + 8 * hw;
                int gn = bn + wc + t * 16 + l15;
                if (gm < M && gn < N) {
                    float vv = fr.f[r];
                    if (bias) vv += bias[gn];
                    if (act == 1) vv = fmaxf(vv, 0.0f);
                    C[(size_t)gm * ldc + gn] = vv;
                }
            }
        }
}

// transpose + convert: in f32 [R][C] -> out f16 [C][R]  (batched over z)
__global__ __launch_bounds__(256)
void transpose_cvt_k(const float* __restrict__ in, _Float16* __restrict__ out,
                     int R, int C, long long inS, long long outS)
{
    __shared__ float tile[32][33];
    in  += (long long)blockIdx.z * inS;
    out += (long long)blockIdx.z * outS;
    int tx = threadIdx.x & 31, ty = threadIdx.x >> 5;
    int c0 = blockIdx.x * 32, r0 = blockIdx.y * 32;
    #pragma unroll
    for (int j = 0; j < 4; ++j) {
        int r = r0 + ty + j * 8, c = c0 + tx;
        if (r < R && c < C) tile[ty + j * 8][tx] = in[(size_t)r * C + c];
    }
    __syncthreads();
    #pragma unroll
    for (int j = 0; j < 4; ++j) {
        int c = c0 + ty + j * 8, r = r0 + tx;
        if (r < R && c < C) out[(size_t)c * R + r] = (_Float16)tile[tx][ty + j * 8];
    }
}

// LayerNorm; writes f32 (outf) and/or f16 (outh)
__global__ __launch_bounds__(256)
void layernorm_k(const float* __restrict__ in, float* __restrict__ outf,
                 _Float16* __restrict__ outh,
                 const float* __restrict__ g, const float* __restrict__ b,
                 int W, long long in_rs, long long out_rs)
{
    __shared__ float red[256];
    long long row = blockIdx.x;
    in += row * in_rs;
    int tid = threadIdx.x;
    float s = 0.f, s2 = 0.f;
    for (int i = tid; i < W; i += 256) { float v = in[i]; s += v; s2 += v * v; }
    red[tid] = s; __syncthreads();
    for (int st = 128; st > 0; st >>= 1) { if (tid < st) red[tid] += red[tid + st]; __syncthreads(); }
    float mean = red[0] / W; __syncthreads();
    red[tid] = s2; __syncthreads();
    for (int st = 128; st > 0; st >>= 1) { if (tid < st) red[tid] += red[tid + st]; __syncthreads(); }
    float var  = red[0] / W - mean * mean;
    float rstd = rsqrtf(var + 1e-5f);
    for (int i = tid; i < W; i += 256) {
        float vv = (in[i] - mean) * rstd * g[i] + b[i];
        if (outf) outf[row * out_rs + i] = vv;
        if (outh) outh[row * out_rs + i] = (_Float16)vv;
    }
}

// causal softmax: read f32 S[h][t][:], write f16 P ; grid = H*T
__global__ __launch_bounds__(256)
void softmax_causal_k(const float* __restrict__ S, _Float16* __restrict__ P, int Tn)
{
    __shared__ float red[256];
    int r = blockIdx.x;
    int hI = r / Tn, t = r % Tn;
    const float* row = S + (size_t)hI * Tn * Tn + (size_t)t * Tn;
    _Float16*   prow = P + (size_t)hI * Tn * Tn + (size_t)t * Tn;
    int tid = threadIdx.x;
    float mx = -3.402823466e38f;
    for (int s = tid; s <= t; s += 256) mx = fmaxf(mx, row[s]);
    red[tid] = mx; __syncthreads();
    for (int st = 128; st > 0; st >>= 1) { if (tid < st) red[tid] = fmaxf(red[tid], red[tid + st]); __syncthreads(); }
    mx = red[0]; __syncthreads();
    float sum = 0.f;
    for (int s = tid; s <= t; s += 256) { float e = __expf(row[s] - mx); sum += e; prow[s] = (_Float16)e; }
    red[tid] = sum; __syncthreads();
    for (int st = 128; st > 0; st >>= 1) { if (tid < st) red[tid] += red[tid + st]; __syncthreads(); }
    float inv = 1.0f / red[0];
    for (int s = tid; s <= t; s += 256) prow[s] = (_Float16)((float)prow[s] * inv);
    for (int s = t + 1 + tid; s < Tn; s += 256) prow[s] = (_Float16)0.0f;
}

__global__ void add_pos_k(float* __restrict__ x, const float* __restrict__ pos,
                          int Tn, int En, long long n)
{
    long long i = (long long)blockIdx.x * blockDim.x + threadIdx.x;
    if (i >= n) return;
    int e = (int)(i % En);
    long long bt = i / En;
    int t = (int)(bt % Tn);
    x[i] += pos[(long long)t * En + e];
}

// ---------------------------------------------------------------------------
extern "C" void kernel_launch(void* const* d_in, const int* in_sizes, int n_in,
                              void* d_out, int out_size, void* d_ws, size_t ws_size,
                              hipStream_t stream)
{
    const int Bb = 8, T = 1024, F = 24, E = 512, H = 8, HS = 64, L = 8, FF = 2048;
    const int MT = Bb * T;
    const float SCALE = 0.125f;

    const float* idx  = (const float*)d_in[0];
    const float* pos  = (const float*)d_in[1];
    const float* w1   = (const float*)d_in[2];
    const float* b1   = (const float*)d_in[3];
    const float* g1   = (const float*)d_in[4];
    const float* be1  = (const float*)d_in[5];
    const float* w2   = (const float*)d_in[6];
    const float* b2   = (const float*)d_in[7];
    const float* g2   = (const float*)d_in[8];
    const float* be2  = (const float*)d_in[9];
    const float* w3   = (const float*)d_in[10];
    const float* b3   = (const float*)d_in[11];
    const float* Wq   = (const float*)d_in[12];
    const float* Wk   = (const float*)d_in[13];
    const float* Wv   = (const float*)d_in[14];
    const float* Wp   = (const float*)d_in[15];
    const float* bp   = (const float*)d_in[16];
    const float* gsa  = (const float*)d_in[17];
    const float* bsa  = (const float*)d_in[18];
    const float* gff  = (const float*)d_in[19];
    const float* bff  = (const float*)d_in[20];
    const float* fw1  = (const float*)d_in[21];
    const float* fb1  = (const float*)d_in[22];
    const float* fw2  = (const float*)d_in[23];
    const float* fb2  = (const float*)d_in[24];
    const float* gf   = (const float*)d_in[25];
    const float* bf   = (const float*)d_in[26];
    const float* lmw  = (const float*)d_in[27];
    const float* lmb  = (const float*)d_in[28];
    float* out = (float*)d_out;

    // --- workspace: f32 region then f16 region ---
    float* wsf = (float*)d_ws;
    size_t o = 0;
    float* x   = wsf + o; o += (size_t)MT * E;
    float* S   = wsf + o; o += (size_t)H * T * T;
    float* t1  = wsf + o; o += (size_t)MT * 100;
    float* t2  = wsf + o; o += (size_t)MT * 200;
    float* xl  = wsf + o; o += (size_t)Bb * E;
    _Float16* wsh = (_Float16*)(wsf + o);
    size_t oh = 0;
    _Float16* hh   = wsh + oh; oh += (size_t)MT * E;
    _Float16* qh   = wsh + oh; oh += (size_t)MT * E;
    _Float16* kh   = wsh + oh; oh += (size_t)MT * E;
    _Float16* atth = wsh + oh; oh += (size_t)MT * E;
    _Float16* vth  = wsh + oh; oh += (size_t)H * HS * MT;   // v transposed (h,d,(b,t))
    _Float16* Ph   = wsh + oh; oh += (size_t)H * T * T;
    _Float16* ffhh = wsh + oh; oh += (size_t)MT * FF;
    _Float16* WqT  = wsh + oh; oh += (size_t)L * H * HS * E;
    _Float16* WkT  = wsh + oh; oh += (size_t)L * H * HS * E;
    _Float16* WvT  = wsh + oh; oh += (size_t)L * H * HS * E;
    _Float16* WpT  = wsh + oh; oh += (size_t)L * E * E;
    _Float16* fw1T = wsh + oh; oh += (size_t)L * FF * E;
    _Float16* fw2T = wsh + oh; oh += (size_t)L * E * FF;

    auto gemm16 = [&](const _Float16* A, const _Float16* Bm, float* Cf, _Float16* Ch,
                      const float* bias, const float* res,
                      int M, int N, int K, int lda, int ldb, int ldc, int ldct,
                      int storeCT, int act, float scale, int nbz, int nb1,
                      long long sA0, long long sA1, long long sB0, long long sB1,
                      long long sC0, long long sC1) {
        dim3 grid(N / BN, M / BM, nbz);
        wmma_gemm_f16nt<<<grid, dim3(NTHR), 0, stream>>>(
            A, Bm, Cf, Ch, bias, res, K, lda, ldb, ldc, ldct, storeCT, act, scale,
            nb1, sA0, sA1, sB0, sB1, sC0, sC1);
    };
    auto gemm32 = [&](const float* A, const float* Bm, float* C, const float* bias,
                      int M, int N, int K, int lda, int ldb, int ldc, int act) {
        dim3 grid((N + BN - 1) / BN, (M + BM - 1) / BM, 1);
        wmma_gemm_f32gen<<<grid, dim3(NTHR), 0, stream>>>(
            A, Bm, C, bias, M, N, K, lda, ldb, ldc, act);
    };
    auto ln = [&](const float* in, float* outf, _Float16* outh,
                  const float* g, const float* b, int rows, int W,
                  long long irs, long long ors) {
        layernorm_k<<<dim3(rows), dim3(256), 0, stream>>>(in, outf, outh, g, b, W, irs, ors);
    };
    auto tcvt = [&](const float* in, _Float16* outp, int R, int C, int nb,
                    long long is, long long os) {
        transpose_cvt_k<<<dim3((C + 31) / 32, (R + 31) / 32, nb), dim3(256), 0, stream>>>(
            in, outp, R, C, is, os);
    };

    // ---- one-time weight transpose+convert (deterministic every call) ------
    tcvt(Wq,  WqT,  E, HS, L * H, (long long)E * HS, (long long)E * HS);
    tcvt(Wk,  WkT,  E, HS, L * H, (long long)E * HS, (long long)E * HS);
    tcvt(Wv,  WvT,  E, HS, L * H, (long long)E * HS, (long long)E * HS);
    tcvt(Wp,  WpT,  E, E,  L,     (long long)E * E,  (long long)E * E);
    tcvt(fw1, fw1T, E, FF, L,     (long long)E * FF, (long long)E * FF);
    tcvt(fw2, fw2T, FF, E, L,     (long long)FF * E, (long long)FF * E);

    // ---- input mapping MLP: F->100->200->E (guarded f32 kernel) ------------
    gemm32(idx, w1, t1, b1, MT, 100, F, F, 100, 100, 1);
    ln(t1, t1, nullptr, g1, be1, MT, 100, 100, 100);
    gemm32(t1, w2, t2, b2, MT, 200, 100, 100, 200, 200, 1);
    ln(t2, t2, nullptr, g2, be2, MT, 200, 200, 200);
    gemm32(t2, w3, x, b3, MT, E, 200, 200, E, E, 0);
    {
        long long n = (long long)MT * E;
        add_pos_k<<<dim3((unsigned)((n + 255) / 256)), dim3(256), 0, stream>>>(x, pos, T, E, n);
    }

    // ---- transformer blocks ------------------------------------------------
    for (int l = 0; l < L; ++l) {
        const _Float16* WqTl = WqT + (size_t)l * H * HS * E;
        const _Float16* WkTl = WkT + (size_t)l * H * HS * E;
        const _Float16* WvTl = WvT + (size_t)l * H * HS * E;
        const _Float16* WpTl = WpT + (size_t)l * E * E;
        const _Float16* fw1Tl = fw1T + (size_t)l * FF * E;
        const _Float16* fw2Tl = fw2T + (size_t)l * E * FF;

        ln(x, nullptr, hh, gsa + (size_t)l * E, bsa + (size_t)l * E, MT, E, E, E);

        // q,k: normal f16 store (b,t,h,d); v: transposed store into vth[h][d][bt]
        gemm16(hh, WqTl, nullptr, qh, nullptr, nullptr, MT, HS, E, E, E, E, 0,
               0, 0, 1.0f, H, 1, 0,0, (long long)HS * E, 0, HS, 0);
        gemm16(hh, WkTl, nullptr, kh, nullptr, nullptr, MT, HS, E, E, E, E, 0,
               0, 0, 1.0f, H, 1, 0,0, (long long)HS * E, 0, HS, 0);
        gemm16(hh, WvTl, nullptr, vth, nullptr, nullptr, MT, HS, E, E, E, 0, MT,
               1, 0, 1.0f, H, 1, 0,0, (long long)HS * E, 0, (long long)HS * MT, 0);

        for (int b = 0; b < Bb; ++b) {
            const _Float16* qb = qh + (size_t)b * T * E;
            const _Float16* kb = kh + (size_t)b * T * E;
            _Float16* attb = atth + (size_t)b * T * E;
            // S[h] = scale * q_bh @ k_bh^T  (both [seq][d] f16 -> NT form)
            gemm16(qb, kb, S, nullptr, nullptr, nullptr, T, T, HS, E, E, T, 0,
                   0, 0, SCALE, H, 1, HS, 0, HS, 0, (long long)T * T, 0);
            softmax_causal_k<<<dim3(H * T), dim3(256), 0, stream>>>(S, Ph, T);
            // att_bh = P @ v_bh  (B = vth rows d, contiguous s)
            gemm16(Ph, vth + (size_t)b * T, nullptr, attb, nullptr, nullptr,
                   T, HS, T, T, MT, E, 0, 0, 0, 1.0f, H, 1,
                   (long long)T * T, 0, (long long)HS * MT, 0, HS, 0);
        }

        // x = x + att @ Wp + bp
        gemm16(atth, WpTl, x, nullptr, bp + (size_t)l * E, x, MT, E, E, E, E, E, 0,
               0, 0, 1.0f, 1, 1, 0,0, 0,0, 0,0);

        // FFN
        ln(x, nullptr, hh, gff + (size_t)l * E, bff + (size_t)l * E, MT, E, E, E);
        gemm16(hh, fw1Tl, nullptr, ffhh, fb1 + (size_t)l * FF, nullptr,
               MT, FF, E, E, E, FF, 0, 0, 2, 1.0f, 1, 1, 0,0, 0,0, 0,0);
        gemm16(ffhh, fw2Tl, x, nullptr, fb2 + (size_t)l * E, x,
               MT, E, FF, FF, FF, E, 0, 0, 0, 1.0f, 1, 1, 0,0, 0,0, 0,0);
    }

    // ---- final LN on last tokens + head ------------------------------------
    ln(x + (size_t)(T - 1) * E, xl, nullptr, gf, bf, Bb, E, (long long)T * E, E);
    gemm32(xl, lmw, out, lmb, Bb, F, E, E, F, F, 0);
}
